// kernel_generated_6_62904091017473
// MI455X (gfx1250) — matrix operand order repaired and checked by reading, not compiled, not run on hardware
//
#include <hip/hip_runtime.h>

// ---------------------------------------------------------------------------
// y[b,l,h,w] = sum_{j,k,i} A[(b,h,w),(j,k,i)] * W[j,k,i,l]
//   A[(b,h,w),(j,k,i)] = mask(0<=w+i-1<7) * S[b,j,k,h,(w+i-2) mod 7]
//   S[b,j,k,h,w] = padH(x[b,j,h+k-1,w]) + padH(x[b,256+j,((h-1)%7)+k-1,w])
// GEMM: C[6272 x 512] = A[6272 x 2304] * Wmat[2304 x 512]  (Wmat = W flat)
// Strategy: bf16 hi/lo split (bf16x3, drop lo*lo), v_wmma_f32_16x16x32_bf16,
// fragment-native staging in workspace so the GEMM hot loop is pure b128 loads.
// ---------------------------------------------------------------------------

typedef __attribute__((ext_vector_type(16))) __bf16 v16bf;
typedef __attribute__((ext_vector_type(8)))  float  v8f;

#define CIN   512
#define NCOL  512
#define MT    392      // 6272 / 16 M-tiles
#define KT    72       // 2304 / 32 K-tiles
#define NT    32       // 512 / 16 N-tiles

// fragment storage: tile(t) -> [lane 0..31][slot 0..15] bf16, contiguous 32B/lane
// A-tile element (lane,slot) -> (mloc = lane&15,
//   kloc = lane<16 ? (s<8 ? s : s+8) : (s<8 ? s+8 : s+16))        [ISA 16b A 16x32]
// B-tile element (lane,slot) -> (n = lane&15, K = same as A kloc)   [ISA 16b B 32x16]

__device__ __forceinline__ void split_bf16(float v, __bf16& hi, __bf16& lo) {
  hi = (__bf16)v;
  lo = (__bf16)(v - (float)hi);
}

__global__ __launch_bounds__(256) void build_a_frags(const float* __restrict__ x,
                                                     __bf16* __restrict__ Ah,
                                                     __bf16* __restrict__ Al) {
  const int bx    = blockIdx.x;          // mtile * KT + ktile
  const int mtile = bx / KT;
  const int ktile = bx - mtile * KT;
  const int tid   = threadIdx.x;
  const int lane  = tid >> 3;            // 0..31
  const int sp    = tid & 7;             // slot pair
  const int mloc  = lane & 15;
  const int m     = mtile * 16 + mloc;   // global row = (b,h,w)
  const int b     = m / 49;
  const int r     = m - b * 49;
  const int h     = r / 7;
  const int w     = r - h * 7;
  const int h2    = (h + 6) % 7;         // (h-1) mod 7
  const size_t base = ((size_t)bx * 32 + lane) * 16;
#pragma unroll
  for (int t = 0; t < 2; ++t) {
    const int s    = sp * 2 + t;
    const int kloc = (lane < 16) ? (s < 8 ? s : s + 8)
                                 : (s < 8 ? s + 8 : s + 16);
    const int col  = ktile * 32 + kloc;  // (j,k,i) flattened
    const int j    = col / 9;
    const int rem  = col - j * 9;
    const int k    = rem / 3;
    const int i    = rem - k * 3;
    float v = 0.f;
    const int wsrc = w + i - 1;
    if (wsrc >= 0 && wsrc < 7) {
      const int wr  = (wsrc + 6) % 7;    // (w+i-2) mod 7
      const int hh1 = h + k - 1;
      if (hh1 >= 0 && hh1 < 7)
        v += x[(((size_t)b * CIN + j) * 7 + hh1) * 7 + wr];
      const int hh2 = h2 + k - 1;
      if (hh2 >= 0 && hh2 < 7)
        v += x[(((size_t)b * CIN + 256 + j) * 7 + hh2) * 7 + wr];
    }
    __bf16 hi, lo;
    split_bf16(v, hi, lo);
    Ah[base + s] = hi;
    Al[base + s] = lo;
  }
}

__global__ __launch_bounds__(256) void build_w_frags(const float* __restrict__ Wsrc,
                                                     __bf16* __restrict__ Bh,
                                                     __bf16* __restrict__ Bl) {
  const int bx    = blockIdx.x;          // ktile * NT + ntile
  const int ktile = bx / NT;
  const int ntile = bx - ktile * NT;
  const int tid   = threadIdx.x;
  const int lane  = tid >> 3;
  const int sp    = tid & 7;
  const int n     = ntile * 16 + (lane & 15);
  const size_t base = ((size_t)bx * 32 + lane) * 16;
#pragma unroll
  for (int t = 0; t < 2; ++t) {
    const int s   = sp * 2 + t;
    const int K   = (lane < 16) ? (s < 8 ? s : s + 8) : (s < 8 ? s + 8 : s + 16);
    const int col = ktile * 32 + K;
    const float v = Wsrc[(size_t)col * NCOL + n];
    __bf16 hi, lo;
    split_bf16(v, hi, lo);
    Bh[base + s] = hi;
    Bl[base + s] = lo;
  }
}

// 8 waves / block; wave tile = 2 M-frags x 4 N-frags = 32x64 fp32 out.
// block tile = 128(M) x 128(N); grid = 49 x 4.
__global__ __launch_bounds__(256) void gemm_bf16x3(const __bf16* __restrict__ Ah,
                                                   const __bf16* __restrict__ Al,
                                                   const __bf16* __restrict__ Bh,
                                                   const __bf16* __restrict__ Bl,
                                                   float* __restrict__ out) {
  const int tid    = threadIdx.x;
  const int lane   = tid & 31;
  const int wv     = tid >> 5;           // 0..7
  const int wvm    = wv & 3;             // 4 waves along M
  const int wvn    = wv >> 2;            // 2 waves along N
  const int mtile0 = blockIdx.x * 8 + wvm * 2;
  const int ntile0 = blockIdx.y * 8 + wvn * 4;

  v8f acc[2][4];
#pragma unroll
  for (int mi = 0; mi < 2; ++mi)
#pragma unroll
    for (int ni = 0; ni < 4; ++ni)
      acc[mi][ni] = (v8f){0.f, 0.f, 0.f, 0.f, 0.f, 0.f, 0.f, 0.f};

  const size_t laneOff = (size_t)lane * 16;

  for (int ktile = 0; ktile < KT; ++ktile) {
    v16bf ah[2], al[2], bh[4], bl[4];
#pragma unroll
    for (int mi = 0; mi < 2; ++mi) {
      const size_t off = ((size_t)((mtile0 + mi) * KT + ktile) * 32) * 16 + laneOff;
      ah[mi] = *(const v16bf*)(Ah + off);
      al[mi] = *(const v16bf*)(Al + off);
      if (ktile + 1 < KT) {               // global_prefetch_b8 of next K-slab
        __builtin_prefetch((const void*)(Ah + off + 512), 0, 0);
        __builtin_prefetch((const void*)(Al + off + 512), 0, 0);
      }
    }
#pragma unroll
    for (int ni = 0; ni < 4; ++ni) {
      const size_t off = ((size_t)(ktile * NT + (ntile0 + ni)) * 32) * 16 + laneOff;
      bh[ni] = *(const v16bf*)(Bh + off);
      bl[ni] = *(const v16bf*)(Bl + off);
    }
#pragma unroll
    for (int mi = 0; mi < 2; ++mi) {
#pragma unroll
      for (int ni = 0; ni < 4; ++ni) {
        acc[mi][ni] = __builtin_amdgcn_wmma_f32_16x16x32_bf16(
            false, ah[mi], false, bh[ni], (short)0, acc[mi][ni], false, false);
        acc[mi][ni] = __builtin_amdgcn_wmma_f32_16x16x32_bf16(
            false, ah[mi], false, bl[ni], (short)0, acc[mi][ni], false, false);
        acc[mi][ni] = __builtin_amdgcn_wmma_f32_16x16x32_bf16(
            false, al[mi], false, bh[ni], (short)0, acc[mi][ni], false, false);
      }
    }
  }

  // C/D 16x16 f32 layout: N = lane&15, M = v + 8*(lane>>4)
  const int nl = lane & 15;
  const int mh = (lane >> 4) << 3;
#pragma unroll
  for (int mi = 0; mi < 2; ++mi) {
    const int mb = (mtile0 + mi) * 16 + mh;
#pragma unroll
    for (int ni = 0; ni < 4; ++ni) {
      const int l = (ntile0 + ni) * 16 + nl;
#pragma unroll
      for (int v = 0; v < 8; ++v) {
        const int mrow = mb + v;
        const int b = mrow / 49;
        const int r = mrow - b * 49;
        out[((size_t)b * NCOL + l) * 49 + r] = acc[mi][ni][v];
      }
    }
  }
}

extern "C" void kernel_launch(void* const* d_in, const int* in_sizes, int n_in,
                              void* d_out, int out_size, void* d_ws, size_t ws_size,
                              hipStream_t stream) {
  const float* x = (const float*)d_in[0];   // [128,512,7,7]
  const float* W = (const float*)d_in[1];   // [256,3,3,512]
  float* out = (float*)d_out;               // [128,512,7,7]

  // workspace layout (needs ~62.6 MB):
  //  Ah: 392*72*1024 bf16 = 28,901,376 B      Al: same
  //  Bh: 72*32*1024 bf16  =  2,359,296 B      Bl: same
  char* ws = (char*)d_ws;
  __bf16* Ah = (__bf16*)(ws);
  __bf16* Al = (__bf16*)(ws + 28901376u);
  __bf16* Bh = (__bf16*)(ws + 57802752u);
  __bf16* Bl = (__bf16*)(ws + 60162048u);

  build_a_frags<<<MT * KT, 256, 0, stream>>>(x, Ah, Al);
  build_w_frags<<<KT * NT, 256, 0, stream>>>(W, Bh, Bl);
  gemm_bf16x3<<<dim3(49, 4), 256, 0, stream>>>(Ah, Al, Bh, Bl, out);
}
